// Attention_18726057410699
// MI455X (gfx1250) — matrix operand order repaired and checked by reading, not compiled, not run on hardware
//
#include <hip/hip_runtime.h>
#include <stdint.h>

typedef __attribute__((ext_vector_type(2)))  float  v2f;
typedef __attribute__((ext_vector_type(8)))  float  v8f;
typedef __attribute__((ext_vector_type(16))) __bf16 v16bf;

#define B_   8
#define N_   192
#define H_   256
#define L2_  128
#define BIN_ 112

// workspace layout (bytes)
#define P_OFF    0u
#define P_BYTES  (B_ * N_ * H_ * 4u)          // 1,572,864
#define S_OFF    (P_OFF + P_BYTES)
#define S_BYTES  (B_ * N_ * N_ * 4u)          // 1,179,648
#define WBP_OFF  (S_OFF + S_BYTES)            // 2,752,512 (32B aligned)
#define WBP_ELEMS (4 * 16 * 32 * 16)          // 32768 bf16 (K padded 112->128)

// ---------------------------------------------------------------------------
// Kernel 0: pack W_bin (112x256 f32) into bf16 WMMA B-fragment order,
// K zero-padded to 128. Slot index: ((kt*16 + nt)*32 + lane)*16 + e
// B-frag (32x16): lane holds col = lane&15, k_local = (lane>>4)*8 + (e>>3)*16 + (e&7).
// ---------------------------------------------------------------------------
__global__ void pack_wbin(const float* __restrict__ Wb, uint16_t* __restrict__ out) {
    int idx  = blockIdx.x * blockDim.x + threadIdx.x;   // 0..32767
    int e    = idx & 15;
    int lane = (idx >> 4) & 31;
    int nt   = (idx >> 9) & 15;
    int kt   = idx >> 13;
    int n = nt * 16 + (lane & 15);
    int k = kt * 32 + (lane >> 4) * 8 + (e >> 3) * 16 + (e & 7);
    float v = (k < BIN_) ? Wb[k * H_ + n] : 0.0f;
    union { __bf16 h; uint16_t u; } cv;
    cv.h = (__bf16)v;
    out[idx] = cv.u;
}

// ---------------------------------------------------------------------------
// Kernel 1: P[b,n,:] = local_feats[b,n,:] @ W_apair   (M=1536, N=256, K=256)
// Exact fp32 via v_wmma_f32_16x16x4_f32, fragments loaded straight from global.
// A-frag 16x4 f32: lane row = lane&15, k = (lane>>4)*2 + {0,1}
// B-frag 4x16 f32: lane col = lane&15, k = (lane>>4)*2 + {0,1}
// ---------------------------------------------------------------------------
__global__ void pgemm(const float* __restrict__ Lf, const float* __restrict__ Wa,
                      float* __restrict__ P) {
    int t    = blockIdx.x * (blockDim.x >> 5) + (threadIdx.x >> 5); // tile 0..1535
    int lane = threadIdx.x & 31;
    int rt   = t >> 4;                 // 0..95  (row tile over flattened b*192+n)
    int ct   = t & 15;                 // 0..15  (col tile over k-out)
    int lr   = lane & 15;
    int hi   = lane >> 4;
    int koff = hi * 2;
    const float* arow = Lf + (size_t)(rt * 16 + lr) * H_;
    int ncol = ct * 16 + lr;
    v8f c = {};
    for (int k0 = 0; k0 < H_; k0 += 4) {
        v2f a, b;
        a.x = arow[k0 + koff];
        a.y = arow[k0 + koff + 1];
        b.x = Wa[(size_t)(k0 + koff) * H_ + ncol];
        b.y = Wa[(size_t)(k0 + koff + 1) * H_ + ncol];
        c = __builtin_amdgcn_wmma_f32_16x16x4_f32(false, a, false, b,
                                                  (short)0, c, false, false);
    }
    int rbase = rt * 16 + hi * 8;
#pragma unroll
    for (int v = 0; v < 8; ++v)
        P[(size_t)(rbase + v) * H_ + ncol] = c[v];
}

// ---------------------------------------------------------------------------
// Kernel 2: fused binary GEMM + attention score, region i<128 & j<128.
// Block = (b,i); wave w handles 16 rows (j0=w*16), all 256 cols, K=112(->128).
// Packed W_bin fragments staged once per block into LDS (64 KB), then the
// hot loop is  ds_load_b128 x2 (B-frag)  ->  v_wmma_f32_16x16x32_bf16.
// Epilogue fuses relu(Q + P_i + P_j + b_bin) . w_att -> sigmoid -> S[b,i,j].
// ---------------------------------------------------------------------------
__global__ void bin_attn(const float* __restrict__ BF, const float* __restrict__ P,
                         const uint16_t* __restrict__ WBP,
                         const float* __restrict__ b_bin, const float* __restrict__ w_att,
                         const float* __restrict__ b_att, float* __restrict__ S) {
    __shared__ alignas(32) uint16_t sWBP[WBP_ELEMS];    // 64 KB
    int bi   = blockIdx.x;             // 0..1023
    int bb   = bi >> 7;
    int ii   = bi & 127;
    int tid  = threadIdx.x;
    int lane = tid & 31;
    int w    = tid >> 5;               // 0..7
    int j0   = w * 16;
    int lr   = lane & 15;
    int hi   = lane >> 4;
    int b8   = hi * 8;

    // Cooperative stage of packed W_bin into LDS (4096 x 16B).
    {
        const uint4* src = (const uint4*)WBP;
        uint4* dst = (uint4*)sWBP;
#pragma unroll
        for (int it = 0; it < 16; ++it)
            dst[tid + it * 256] = src[tid + it * 256];
    }

    // Build the 4 A-fragments (16 rows x 128 k, bf16) for this wave.
    // A-frag 16x32 bf16: lane row = lane&15; element e -> k = b8 + (e>>3)*16 + (e&7)
    const float* arow = BF + (size_t)((bb * L2_ + ii) * L2_ + (j0 + lr)) * BIN_;
    v16bf afr[4];
#pragma unroll
    for (int kt = 0; kt < 4; ++kt) {
        float r[16];
        const float4* p1 = (const float4*)(arow + kt * 32 + b8);
        float4 a0 = p1[0], a1 = p1[1];
        r[0] = a0.x; r[1] = a0.y; r[2] = a0.z; r[3] = a0.w;
        r[4] = a1.x; r[5] = a1.y; r[6] = a1.z; r[7] = a1.w;
        if (kt < 3) {
            const float4* p2 = (const float4*)(arow + kt * 32 + b8 + 16);
            float4 c0 = p2[0], c1 = p2[1];
            r[8]  = c0.x; r[9]  = c0.y; r[10] = c0.z; r[11] = c0.w;
            r[12] = c1.x; r[13] = c1.y; r[14] = c1.z; r[15] = c1.w;
        } else {
#pragma unroll
            for (int e = 8; e < 16; ++e) r[e] = 0.0f;   // K pad 112->128
        }
#pragma unroll
        for (int e = 0; e < 16; ++e) afr[kt][e] = (__bf16)r[e];
    }

    __syncthreads();

    float s[8] = {0.f, 0.f, 0.f, 0.f, 0.f, 0.f, 0.f, 0.f};
    float batt = b_att[0];
    const float* Prow_i = P + (size_t)(bb * N_ + ii) * H_;
    const v16bf* wbl = (const v16bf*)sWBP;

    for (int nt = 0; nt < 16; ++nt) {
        v8f c = {};
#pragma unroll
        for (int kt = 0; kt < 4; ++kt) {
            v16bf bf = wbl[(kt * 16 + nt) * 32 + lane];
            c = __builtin_amdgcn_wmma_f32_16x16x32_bf16(false, afr[kt], false, bf,
                                                        (short)0, c, false, false);
        }
        int col   = nt * 16 + lr;
        float pi  = Prow_i[col];
        float bbn = b_bin[col];
        float wa  = w_att[col];
        const float* Pj = P + (size_t)(bb * N_ + j0 + b8) * H_ + col;
#pragma unroll
        for (int v = 0; v < 8; ++v) {
            float val = c[v] + pi + Pj[(size_t)v * H_] + bbn;
            val = fmaxf(val, 0.0f);
            s[v] += val * wa;
        }
    }
    // Reduce over the 16 lanes sharing each D-row (cols are striped mod 16).
#pragma unroll
    for (int v = 0; v < 8; ++v) {
        s[v] += __shfl_xor(s[v], 1);
        s[v] += __shfl_xor(s[v], 2);
        s[v] += __shfl_xor(s[v], 4);
        s[v] += __shfl_xor(s[v], 8);
    }
    if (lr < 8) {
        int j = j0 + b8 + lr;          // row = v + hi*8, v = lr
        float x = s[lr] + batt;
        float sc = 1.0f / (1.0f + __expf(-x));
        S[(size_t)(bb * N_ + ii) * N_ + j] = sc;
    }
}

// ---------------------------------------------------------------------------
// Kernel 3: scores where the binary term is zero-padded (i>=128 or j>=128).
// Block = (b,i); thread j. P_i and w_att staged in LDS; P_j streams from L2.
// ---------------------------------------------------------------------------
__global__ void outer_score(const float* __restrict__ P, const float* __restrict__ w_att,
                            const float* __restrict__ b_att, float* __restrict__ S) {
    __shared__ float spi[H_];
    __shared__ float swa[H_];
    int bb = blockIdx.x / N_;
    int ii = blockIdx.x % N_;
    int j  = threadIdx.x;              // 0..191
    for (int k = threadIdx.x; k < H_; k += blockDim.x) {
        spi[k] = P[(size_t)(bb * N_ + ii) * H_ + k];
        swa[k] = w_att[k];
    }
    __syncthreads();
    if (ii < L2_ && j < L2_) return;   // inner region handled by bin_attn
    const float* Pj = P + (size_t)(bb * N_ + j) * H_;
    float acc = 0.0f;
    for (int k = 0; k < H_; ++k)
        acc += fmaxf(spi[k] + Pj[k], 0.0f) * swa[k];
    float x = acc + b_att[0];
    S[(size_t)(bb * N_ + ii) * N_ + j] = 1.0f / (1.0f + __expf(-x));
}

// ---------------------------------------------------------------------------
// Kernel 4: global_feats[b] = S[b] (192x192) @ local_feats[b] (192x256),
// exact fp32 via v_wmma_f32_16x16x4_f32.
// ---------------------------------------------------------------------------
__global__ void out_gemm(const float* __restrict__ S, const float* __restrict__ Lf,
                         float* __restrict__ O) {
    int t    = blockIdx.x * (blockDim.x >> 5) + (threadIdx.x >> 5); // 0..1535
    int lane = threadIdx.x & 31;
    int bb = t / 192;
    int r  = t % 192;
    int rt = r >> 4;                   // 0..11
    int ct = r & 15;                   // 0..15
    int lr = lane & 15;
    int hi = lane >> 4;
    int koff = hi * 2;
    const float* Arow  = S  + (size_t)(bb * N_ + rt * 16 + lr) * N_;
    const float* Bbase = Lf + (size_t)bb * N_ * H_;
    int ncol = ct * 16 + lr;
    v8f c = {};
    for (int k0 = 0; k0 < N_; k0 += 4) {
        v2f a, b;
        a.x = Arow[k0 + koff];
        a.y = Arow[k0 + koff + 1];
        b.x = Bbase[(size_t)(k0 + koff) * H_ + ncol];
        b.y = Bbase[(size_t)(k0 + koff + 1) * H_ + ncol];
        c = __builtin_amdgcn_wmma_f32_16x16x4_f32(false, a, false, b,
                                                  (short)0, c, false, false);
    }
    int rbase = rt * 16 + hi * 8;
#pragma unroll
    for (int v = 0; v < 8; ++v)
        O[(size_t)(bb * N_ + rbase + v) * H_ + ncol] = c[v];
}

// ---------------------------------------------------------------------------
extern "C" void kernel_launch(void* const* d_in, const int* in_sizes, int n_in,
                              void* d_out, int out_size, void* d_ws, size_t ws_size,
                              hipStream_t stream) {
    const float* Lf   = (const float*)d_in[0];   // [8,192,256]
    const float* BF   = (const float*)d_in[1];   // [8,128,128,112]
    // d_in[2] sparse_idx: unused by the reference
    const float* Wa   = (const float*)d_in[3];   // [256,256]
    const float* Wb   = (const float*)d_in[4];   // [112,256]
    const float* bbin = (const float*)d_in[5];   // [256]
    const float* watt = (const float*)d_in[6];   // [256]
    const float* batt = (const float*)d_in[7];   // [1]
    float* out = (float*)d_out;                  // [8,192,256]

    char* ws = (char*)d_ws;
    float*    P   = (float*)(ws + P_OFF);        // [8*192,256]
    float*    S   = (float*)(ws + S_OFF);        // [8,192,192] sigmoid scores
    uint16_t* WBP = (uint16_t*)(ws + WBP_OFF);   // packed bf16 W_bin fragments

    pack_wbin  <<<WBP_ELEMS / 256, 256, 0, stream>>>(Wb, WBP);
    pgemm      <<<192,  256, 0, stream>>>(Lf, Wa, P);
    bin_attn   <<<1024, 256, 0, stream>>>(BF, P, WBP, bbin, watt, batt, S);
    outer_score<<<B_ * N_, 192, 0, stream>>>(P, watt, batt, S);
    out_gemm   <<<192,  256, 0, stream>>>(S, Lf, out);
}
